// SimpleLiDARBEVEncoder_73813307949093
// MI455X (gfx1250) — matrix operand order repaired and checked by reading, not compiled, not run on hardware
//
#include <hip/hip_runtime.h>
#include <hip/hip_bf16.h>

// ---------------------------------------------------------------------------
// Types for CDNA5 WMMA
// ---------------------------------------------------------------------------
typedef __attribute__((ext_vector_type(16))) __bf16 v16bf;
typedef __attribute__((ext_vector_type(8)))  float  v8f;

#define H_ 200
#define W_ 200
#define B_ 4
#define NPTS_ 100000
#define PADW 202
#define EPS_ 1e-5f

// ---------------------------------------------------------------------------
// Prep: fold BN (eval) into per-channel scale/bias:
//   y = (x + b - m) * g/sqrt(v+eps) + be  ->  y = x*scale + bias
// ---------------------------------------------------------------------------
__global__ void fold_bn_kernel(const float* __restrict__ g, const float* __restrict__ be,
                               const float* __restrict__ m, const float* __restrict__ v,
                               const float* __restrict__ b,
                               float* __restrict__ scale, float* __restrict__ bias, int n) {
    int i = blockIdx.x * blockDim.x + threadIdx.x;
    if (i < n) {
        float s = g[i] * rsqrtf(v[i] + EPS_);
        scale[i] = s;
        bias[i] = (b[i] - m[i]) * s + be[i];
    }
}

// Transpose conv weights (3,3,I,O) f32 -> (9,O,I) bf16 (K-contiguous per column)
__global__ void transpose_w_kernel(const float* __restrict__ w, __hip_bfloat16* __restrict__ wT,
                                   int I, int O) {
    int idx = blockIdx.x * blockDim.x + threadIdx.x;
    int total = 9 * I * O;
    if (idx >= total) return;
    int o = idx % O;
    int rem = idx / O;
    int ci = rem % I;
    int tap = rem / I;
    wT[((size_t)tap * O + o) * I + ci] = __float2bfloat16(w[((size_t)tap * I + ci) * O + o]);
}

// ---------------------------------------------------------------------------
// Pillar MLP + scatter-max.  grid must be zero-initialized.
// Scatter-max done as int atomicMax on float bits: correct for max(f..., 0)
// because negative floats pun to negative ints (never beat 0) and positives
// order identically.
// ---------------------------------------------------------------------------
__global__ __launch_bounds__(256)
void mlp_scatter_kernel(const float* __restrict__ points,
                        const float* __restrict__ w1,     // (64,5)
                        const float* __restrict__ s1t1,   // [0..63]=scale, [64..127]=bias
                        const float* __restrict__ w2,     // (128,64)
                        const float* __restrict__ b2,     // (128)
                        float* __restrict__ grid) {       // (B,200,200,128) zero-init
    __shared__ float sw1[64 * 5];
    __shared__ float ss[64], st[64];
    __shared__ float sw2[128 * 64];
    __shared__ float sb2[128];
    for (int i = threadIdx.x; i < 320; i += 256) sw1[i] = w1[i];
    for (int i = threadIdx.x; i < 64; i += 256) { ss[i] = s1t1[i]; st[i] = s1t1[64 + i]; }
    for (int i = threadIdx.x; i < 128 * 64; i += 256) sw2[i] = w2[i];
    for (int i = threadIdx.x; i < 128; i += 256) sb2[i] = b2[i];
    __syncthreads();

    int idx = blockIdx.x * 256 + threadIdx.x;
    if (idx >= B_ * NPTS_) return;

    const float* p = points + (size_t)idx * 5;
    float x0 = p[0], x1 = p[1], x2 = p[2], x3 = p[3], x4 = p[4];

    int gx = (int)floorf((x0 + 51.2f) * (1.0f / 0.512f));
    int gy = (int)floorf((x1 + 51.2f) * (1.0f / 0.512f));
    bool valid = ((x0 != 0.0f) || (x1 != 0.0f)) &&
                 (gx >= 0) && (gx < W_) && (gy >= 0) && (gy < H_);
    if (!valid) return;

    float h[64];
#pragma unroll
    for (int c = 0; c < 64; ++c) {
        const float* wr = sw1 + c * 5;
        float a = wr[0] * x0 + wr[1] * x1 + wr[2] * x2 + wr[3] * x3 + wr[4] * x4;
        h[c] = fmaxf(a * ss[c] + st[c], 0.0f);
    }

    int b = idx / NPTS_;
    size_t base = ((size_t)b * (H_ * W_) + (size_t)gy * W_ + gx) * 128;
    int* gi = (int*)grid;
    for (int o = 0; o < 128; ++o) {
        float a = sb2[o];
        const float* wr = sw2 + o * 64;
#pragma unroll
        for (int c = 0; c < 64; ++c) a += h[c] * wr[c];
        atomicMax(gi + base + o, __float_as_int(a));
    }
}

// ---------------------------------------------------------------------------
// Convert f32 grid (B,200,200,128) -> padded bf16 act (B,202,202,128)
// ---------------------------------------------------------------------------
__global__ __launch_bounds__(256)
void grid_to_bf16_kernel(const float* __restrict__ grid, __hip_bfloat16* __restrict__ act) {
    int idx = blockIdx.x * 256 + threadIdx.x;
    int total = B_ * H_ * W_ * 32;  // 4 channels per thread
    if (idx >= total) return;
    int c4 = idx & 31;
    int pix = idx >> 5;
    int x = pix % W_;
    int rem = pix / W_;
    int y = rem % H_;
    int b = rem / H_;
    float4 f = *(const float4*)(grid + (size_t)pix * 128 + c4 * 4);
    union { __hip_bfloat16 h[4]; uint2 u; } o;
    o.h[0] = __float2bfloat16(f.x);
    o.h[1] = __float2bfloat16(f.y);
    o.h[2] = __float2bfloat16(f.z);
    o.h[3] = __float2bfloat16(f.w);
    size_t dst = (((size_t)b * PADW + (y + 1)) * PADW + (x + 1)) * 128 + c4 * 4;
    *(uint2*)(act + dst) = o.u;
}

// ---------------------------------------------------------------------------
// 3x3 SAME conv via implicit GEMM with v_wmma_f32_16x16x32_bf16.
//   act_in : padded bf16 (B, 202, 202, CIN), zero border
//   wT     : bf16 (9, COUT, CIN)   (column = output channel, K-contiguous)
//   epilogue: y = relu(acc*scale[c] + bias[c])
//
// Workgroup: 256 threads = 8 waves; WG tile = 4 rows x 16 cols (64 px) x 128 ch.
// Each wave: 2x2 register block = 32M (2 rows x 16 cols) x 32N,
// 4 independent accumulators -> A/B fragments reused twice each
// (2 loads per WMMA) and 4 interleaved accumulation chains on the XDL pipe.
// ---------------------------------------------------------------------------
template <int CIN, bool FINAL>
__global__ __launch_bounds__(256)
void conv3x3_wmma_kernel(const __hip_bfloat16* __restrict__ act_in,
                         const __hip_bfloat16* __restrict__ wT,
                         const float* __restrict__ scale,
                         const float* __restrict__ bias,
                         __hip_bfloat16* __restrict__ act_out,  // padded (B,202,202,COUT)
                         float* __restrict__ out_nchw,          // (B,COUT,200,200)
                         int COUT) {
    extern __shared__ char smem_raw[];
    __hip_bfloat16* lds = (__hip_bfloat16*)smem_raw;  // patch (6, 18, CIN)

    const int tid = threadIdx.x;
    const int lane = tid & 31;
    const int wave = tid >> 5;
    const int mi = wave & 1;    // row-pair within WG tile (rows 2*mi, 2*mi+1)
    const int ni = wave >> 1;   // N group of 32 (0..3)
    const int x0 = blockIdx.x * 16;   // 13 tiles cover 200 (96% util)
    const int y0 = blockIdx.y * 4;    // 50 tiles, exact
    const int nTiles = COUT >> 7;     // 128 output channels per WG
    const int b = blockIdx.z / nTiles;
    const int nb = (blockIdx.z % nTiles) * 128;

    // Cooperative stage: padded rows y0..y0+5, cols x0..x0+17, all CIN
    const int CPV = CIN / 8;  // 16B chunks per pixel
    for (int i = tid; i < 6 * 18 * CPV; i += 256) {
        int ci8 = i % CPV;
        int rc = i / CPV;
        int c = rc % 18;
        int r = rc / 18;
        uint4 v = make_uint4(0, 0, 0, 0);
        int col = x0 + c;
        if (col < PADW) {
            v = *(const uint4*)(act_in + ((((size_t)b * PADW + (y0 + r)) * PADW + col) * CIN) + ci8 * 8);
        }
        *(uint4*)(lds + ((size_t)(r * 18 + c) * CIN) + ci8 * 8) = v;
    }
    __syncthreads();

    v8f acc00 = {0.f, 0.f, 0.f, 0.f, 0.f, 0.f, 0.f, 0.f};
    v8f acc01 = acc00, acc10 = acc00, acc11 = acc00;

    const int m = lane & 15;       // A col-offset / B column within 16
    const int hiHalf = lane >> 4;  // lane half selects K sub-range
    const int r0 = mi * 2;         // first image row of this wave's M block
    const int nc0 = nb + ni * 32 + m;  // B column of fragment j=0

#pragma unroll
    for (int ky = 0; ky < 3; ++ky) {
#pragma unroll
        for (int kx = 0; kx < 3; ++kx) {
            const __hip_bfloat16* arow0 = lds + (size_t)((r0 + ky) * 18 + m + kx) * CIN;
            const __hip_bfloat16* arow1 = arow0 + (size_t)18 * CIN;  // next image row
            const __hip_bfloat16* brow0 = wT + ((size_t)(ky * 3 + kx) * COUT + nc0) * CIN;
            const __hip_bfloat16* brow1 = brow0 + (size_t)16 * CIN;  // +16 channels
#pragma unroll
            for (int kc = 0; kc < CIN; kc += 32) {
                // A fragments (16x32 bf16): lanes 0-15 hold K{0-7,16-23},
                // lanes 16-31 hold K{8-15,24-31} for the same GEMM row.
                union { uint4 u[2]; v16bf v; } A0, A1, B0, B1;
                const __hip_bfloat16* ap0 = arow0 + kc + hiHalf * 8;
                A0.u[0] = *(const uint4*)(ap0);
                A0.u[1] = *(const uint4*)(ap0 + 16);
                const __hip_bfloat16* ap1 = arow1 + kc + hiHalf * 8;
                A1.u[0] = *(const uint4*)(ap1);
                A1.u[1] = *(const uint4*)(ap1 + 16);
                // B fragments (32x16 bf16): lane = column N; lanes 0-15 hold
                // K{0-7,16-23}, lanes 16-31 hold K{8-15,24-31}.
                const __hip_bfloat16* bp0 = brow0 + kc + hiHalf * 8;
                B0.u[0] = *(const uint4*)(bp0);
                B0.u[1] = *(const uint4*)(bp0 + 16);
                const __hip_bfloat16* bp1 = brow1 + kc + hiHalf * 8;
                B1.u[0] = *(const uint4*)(bp1);
                B1.u[1] = *(const uint4*)(bp1 + 16);
                // 4 independent accumulation chains
                acc00 = __builtin_amdgcn_wmma_f32_16x16x32_bf16(
                    false, A0.v, false, B0.v, (short)0, acc00, false, false);
                acc10 = __builtin_amdgcn_wmma_f32_16x16x32_bf16(
                    false, A1.v, false, B0.v, (short)0, acc10, false, false);
                acc01 = __builtin_amdgcn_wmma_f32_16x16x32_bf16(
                    false, A0.v, false, B1.v, (short)0, acc01, false, false);
                acc11 = __builtin_amdgcn_wmma_f32_16x16x32_bf16(
                    false, A1.v, false, B1.v, (short)0, acc11, false, false);
            }
        }
    }

    // Epilogue: D layout — VGPR r: M = r + 8*hiHalf (col offset), N = lane&15
    const float s0 = scale[nc0], t0 = bias[nc0];
    const float s1 = scale[nc0 + 16], t1 = bias[nc0 + 16];
#pragma unroll
    for (int i = 0; i < 2; ++i) {
        int row = y0 + r0 + i;
        const v8f* accs0 = (i == 0) ? &acc00 : &acc10;
        const v8f* accs1 = (i == 0) ? &acc01 : &acc11;
#pragma unroll
        for (int r = 0; r < 8; ++r) {
            int px = x0 + r + hiHalf * 8;
            if (px < W_) {
                float v0 = fmaxf((*accs0)[r] * s0 + t0, 0.0f);
                float v1 = fmaxf((*accs1)[r] * s1 + t1, 0.0f);
                if (FINAL) {
                    out_nchw[(((size_t)b * COUT + nc0) * H_ + row) * W_ + px] = v0;
                    out_nchw[(((size_t)b * COUT + nc0 + 16) * H_ + row) * W_ + px] = v1;
                } else {
                    size_t base = (((size_t)b * PADW + (row + 1)) * PADW + (px + 1)) * COUT;
                    act_out[base + nc0] = __float2bfloat16(v0);
                    act_out[base + nc0 + 16] = __float2bfloat16(v1);
                }
            }
        }
    }
}

// ---------------------------------------------------------------------------
// Workspace layout (bytes)
// ---------------------------------------------------------------------------
static const size_t GRID_BYTES  = (size_t)B_ * H_ * W_ * 128 * 4;        // 81,920,000
static const size_t ACT2_BYTES  = (size_t)B_ * PADW * PADW * 256 * 2;    // 83,566,592
static const size_t REGA_BYTES  = ACT2_BYTES;                            // grid & act2 share
static const size_t ACT01_BYTES = (size_t)B_ * PADW * PADW * 128 * 2;    // 41,783,296
static const size_t OFF_ACT0 = REGA_BYTES;
static const size_t OFF_ACT1 = OFF_ACT0 + ACT01_BYTES;
static const size_t OFF_WT1  = OFF_ACT1 + ACT01_BYTES;
static const size_t WT1_B = (size_t)9 * 128 * 128 * 2;
static const size_t OFF_WT2 = OFF_WT1 + WT1_B;
static const size_t WT2_B = (size_t)9 * 128 * 256 * 2;
static const size_t OFF_WT3 = OFF_WT2 + WT2_B;
static const size_t WT3_B = (size_t)9 * 256 * 256 * 2;
static const size_t OFF_SC = OFF_WT3 + WT3_B;

extern "C" void kernel_launch(void* const* d_in, const int* in_sizes, int n_in,
                              void* d_out, int out_size, void* d_ws, size_t ws_size,
                              hipStream_t stream) {
    const float* points = (const float*)d_in[0];
    const float* w1  = (const float*)d_in[1];
    const float* b1  = (const float*)d_in[2];
    const float* g1  = (const float*)d_in[3];
    const float* be1 = (const float*)d_in[4];
    const float* m1  = (const float*)d_in[5];
    const float* v1  = (const float*)d_in[6];
    const float* w2  = (const float*)d_in[7];
    const float* b2  = (const float*)d_in[8];
    const float* cw1 = (const float*)d_in[9];
    const float* cb1 = (const float*)d_in[10];
    const float* cg1 = (const float*)d_in[11];
    const float* cbe1 = (const float*)d_in[12];
    const float* cm1 = (const float*)d_in[13];
    const float* cv1 = (const float*)d_in[14];
    const float* cw2 = (const float*)d_in[15];
    const float* cb2 = (const float*)d_in[16];
    const float* cg2 = (const float*)d_in[17];
    const float* cbe2 = (const float*)d_in[18];
    const float* cm2 = (const float*)d_in[19];
    const float* cv2 = (const float*)d_in[20];
    const float* cw3 = (const float*)d_in[21];
    const float* cb3 = (const float*)d_in[22];
    const float* cg3 = (const float*)d_in[23];
    const float* cbe3 = (const float*)d_in[24];
    const float* cm3 = (const float*)d_in[25];
    const float* cv3 = (const float*)d_in[26];

    char* ws = (char*)d_ws;
    float* grid = (float*)(ws);                               // region A (f32)
    __hip_bfloat16* act2 = (__hip_bfloat16*)(ws);             // region A reused
    __hip_bfloat16* act0 = (__hip_bfloat16*)(ws + OFF_ACT0);
    __hip_bfloat16* act1 = (__hip_bfloat16*)(ws + OFF_ACT1);
    __hip_bfloat16* wT1 = (__hip_bfloat16*)(ws + OFF_WT1);
    __hip_bfloat16* wT2 = (__hip_bfloat16*)(ws + OFF_WT2);
    __hip_bfloat16* wT3 = (__hip_bfloat16*)(ws + OFF_WT3);
    float* s1t1 = (float*)(ws + OFF_SC);        // 128 floats
    float* sc1 = s1t1 + 128;                    // 128
    float* bi1 = sc1 + 128;                     // 128
    float* sc2 = bi1 + 128;                     // 256
    float* bi2 = sc2 + 256;                     // 256
    float* sc3 = bi2 + 256;                     // 256
    float* bi3 = sc3 + 256;                     // 256

    // --- prep: fold BNs, transpose weights to bf16 (9,O,I) ---
    fold_bn_kernel<<<1, 64, 0, stream>>>(g1, be1, m1, v1, b1, s1t1, s1t1 + 64, 64);
    fold_bn_kernel<<<1, 128, 0, stream>>>(cg1, cbe1, cm1, cv1, cb1, sc1, bi1, 128);
    fold_bn_kernel<<<1, 256, 0, stream>>>(cg2, cbe2, cm2, cv2, cb2, sc2, bi2, 256);
    fold_bn_kernel<<<1, 256, 0, stream>>>(cg3, cbe3, cm3, cv3, cb3, sc3, bi3, 256);
    transpose_w_kernel<<<(9 * 128 * 128 + 255) / 256, 256, 0, stream>>>(cw1, wT1, 128, 128);
    transpose_w_kernel<<<(9 * 128 * 256 + 255) / 256, 256, 0, stream>>>(cw2, wT2, 128, 256);
    transpose_w_kernel<<<(9 * 256 * 256 + 255) / 256, 256, 0, stream>>>(cw3, wT3, 256, 256);

    // --- zero grid (scatter-max init) + act borders ---
    hipMemsetAsync(grid, 0, GRID_BYTES, stream);
    hipMemsetAsync(act0, 0, ACT01_BYTES, stream);
    hipMemsetAsync(act1, 0, ACT01_BYTES, stream);

    // --- pillar MLP + scatter-max ---
    mlp_scatter_kernel<<<(B_ * NPTS_ + 255) / 256, 256, 0, stream>>>(
        points, w1, s1t1, w2, b2, grid);

    // --- grid f32 -> padded bf16 act0 ---
    grid_to_bf16_kernel<<<(B_ * H_ * W_ * 32 + 255) / 256, 256, 0, stream>>>(grid, act0);

    // --- conv1: 128 -> 128 (act0 -> act1) ---
    conv3x3_wmma_kernel<128, false><<<dim3(13, 50, B_ * 1), 256, 6 * 18 * 128 * 2, stream>>>(
        act0, wT1, sc1, bi1, act1, nullptr, 128);

    // region A is done serving as f32 grid; re-zero for act2's borders
    hipMemsetAsync(ws, 0, ACT2_BYTES, stream);

    // --- conv2: 128 -> 256 (act1 -> act2) ---
    conv3x3_wmma_kernel<128, false><<<dim3(13, 50, B_ * 2), 256, 6 * 18 * 128 * 2, stream>>>(
        act1, wT2, sc2, bi2, act2, nullptr, 256);

    // --- conv3: 256 -> 256 (act2 -> d_out, f32 NCHW) ---
    conv3x3_wmma_kernel<256, true><<<dim3(13, 50, B_ * 2), 256, 6 * 18 * 256 * 2, stream>>>(
        act2, wT3, sc3, bi3, nullptr, (float*)d_out, 256);

    (void)in_sizes; (void)n_in; (void)out_size; (void)ws_size;
}
